// PointNetFeaturePropagation_46119358825118
// MI455X (gfx1250) — matrix operand order repaired and checked by reading, not compiled, not run on hardware
//
#include <hip/hip_runtime.h>
#include <hip/hip_bf16.h>
#include <stdint.h>

typedef _Float16 v16h __attribute__((ext_vector_type(16)));
typedef float    v8f  __attribute__((ext_vector_type(8)));

static constexpr int B = 16, N = 4096, M = 1024, C1 = 256, C2 = 256, H = 256;
static constexpr int K1 = C1 + C2;          // 512
static constexpr float EPS_I  = 1e-8f;
static constexpr float EPS_BN = 1e-5f;

// ---------------------------------------------------------------- 3-NN + weights
__global__ __launch_bounds__(256) void knn_kernel(const float* __restrict__ pc1,
                                                  const float* __restrict__ pc2,
                                                  int* __restrict__ idx3,
                                                  float* __restrict__ w3) {
    __shared__ float sx[M], sy[M], sz[M];
    const int b = blockIdx.y;
    const float* p2 = pc2 + (size_t)b * 3 * M;
    for (int i = threadIdx.x; i < M; i += 256) {
        sx[i] = p2[i]; sy[i] = p2[M + i]; sz[i] = p2[2 * M + i];
    }
    __syncthreads();
    const int n = blockIdx.x * 256 + threadIdx.x;
    const float* p1 = pc1 + (size_t)b * 3 * N;
    const float x = p1[n], y = p1[N + n], z = p1[2 * N + n];
    float d0 = 3.4e38f, d1 = 3.4e38f, d2 = 3.4e38f;
    int   i0 = 0, i1 = 0, i2 = 0;
    for (int m = 0; m < M; ++m) {
        float dx = x - sx[m], dy = y - sy[m], dz = z - sz[m];
        float d = dx * dx + dy * dy + dz * dz;
        if (d < d2) {
            if (d < d1) {
                d2 = d1; i2 = i1;
                if (d < d0) { d1 = d0; i1 = i0; d0 = d; i0 = m; }
                else        { d1 = d;  i1 = m; }
            } else { d2 = d; i2 = m; }
        }
    }
    float r0 = 1.f / (d0 + EPS_I), r1 = 1.f / (d1 + EPS_I), r2 = 1.f / (d2 + EPS_I);
    float inv = 1.f / (r0 + r1 + r2);
    size_t base = ((size_t)b * N + n) * 3;
    idx3[base] = i0; idx3[base + 1] = i1; idx3[base + 2] = i2;
    w3[base] = r0 * inv; w3[base + 1] = r1 * inv; w3[base + 2] = r2 * inv;
}

// ------------------------------------------------- feat2 [B,C,M] -> f2t [B,M,C] f32
__global__ __launch_bounds__(256) void f2_transpose(const float* __restrict__ feat2,
                                                    float* __restrict__ f2t) {
    __shared__ float t[32][33];
    const int b = blockIdx.z;
    const int m0 = blockIdx.x * 32, c0 = blockIdx.y * 32;
    const int tx = threadIdx.x, ty = threadIdx.y;
    for (int r = 0; r < 4; ++r)
        t[ty + 8 * r][tx] = feat2[((size_t)b * C2 + c0 + ty + 8 * r) * M + m0 + tx];
    __syncthreads();
    for (int r = 0; r < 4; ++r)
        f2t[((size_t)b * M + m0 + ty + 8 * r) * C2 + c0 + tx] = t[tx][ty + 8 * r];
}

// ------------------------------------- feat1 [B,C,N] -> Xh [B][N][K1] f16 (c < 256)
__global__ __launch_bounds__(256) void f1_transpose(const float* __restrict__ feat1,
                                                    _Float16* __restrict__ Xh) {
    __shared__ float t[32][33];
    const int b = blockIdx.z;
    const int n0 = blockIdx.x * 32, c0 = blockIdx.y * 32;
    const int tx = threadIdx.x, ty = threadIdx.y;
    for (int r = 0; r < 4; ++r)
        t[ty + 8 * r][tx] = feat1[((size_t)b * C1 + c0 + ty + 8 * r) * N + n0 + tx];
    __syncthreads();
    for (int r = 0; r < 4; ++r)
        Xh[((size_t)b * N + n0 + ty + 8 * r) * K1 + c0 + tx] = (_Float16)t[tx][ty + 8 * r];
}

// ----------------------------- interpolation: Xh upper half (c in [256,512)) f16
struct h2v { _Float16 x, y; };
__global__ __launch_bounds__(256) void interp_kernel(const float* __restrict__ f2t,
                                                     const int* __restrict__ idx3,
                                                     const float* __restrict__ w3,
                                                     _Float16* __restrict__ Xh) {
    const int b = blockIdx.y;
    const int wv = threadIdx.x >> 5, lane = threadIdx.x & 31;
    const int n = blockIdx.x * 8 + wv;
    const size_t nb = ((size_t)b * N + n) * 3;
    const int i0 = idx3[nb], i1 = idx3[nb + 1], i2 = idx3[nb + 2];
    const float w0 = w3[nb], w1 = w3[nb + 1], w2 = w3[nb + 2];
    const float* r0 = f2t + ((size_t)b * M + i0) * C2;
    const float* r1 = f2t + ((size_t)b * M + i1) * C2;
    const float* r2 = f2t + ((size_t)b * M + i2) * C2;
    _Float16* out = Xh + ((size_t)b * N + n) * K1 + C1;
#pragma unroll
    for (int j = 0; j < 4; ++j) {
        const int c = 2 * lane + 64 * j;
        float2 a  = *(const float2*)(r0 + c);
        float2 bb = *(const float2*)(r1 + c);
        float2 cc = *(const float2*)(r2 + c);
        h2v o;
        o.x = (_Float16)(w0 * a.x + w1 * bb.x + w2 * cc.x);
        o.y = (_Float16)(w0 * a.y + w1 * bb.y + w2 * cc.y);
        *(h2v*)(out + c) = o;
    }
}

// --------------------------------------------------------------- WMMA GEMM + stats
// h[b,o,n] = sum_c W[o,c]*X[b,n,c] + bias[o];  also atomic sum / sum^2 per channel.
template <int K>
__global__ __launch_bounds__(256) void gemm_kernel(const _Float16* __restrict__ X,
                                                   const float* __restrict__ Wg,
                                                   const float* __restrict__ bias,
                                                   float* __restrict__ hpre,
                                                   float* __restrict__ sum,
                                                   float* __restrict__ sq) {
    __shared__ _Float16 As[32 * K];
    const int b = blockIdx.z;
    const int obase = blockIdx.y * 32;
    const int nblk = blockIdx.x * 256;
    const int tid = threadIdx.x;
    for (int i = tid; i < 32 * K; i += 256) {
        const int r = i / K, c = i % K;             // K is a power of two -> shifts
        As[i] = (_Float16)Wg[(size_t)(obase + r) * K + c];
    }
    __syncthreads();

    const int wv = tid >> 5, lane = tid & 31;
    const int wo = wv & 1, wn = wv >> 1;
    const int laneLo = lane & 15, laneHi = (lane >> 4) & 1;
    const int nwb = nblk + wn * 64;

    v8f acc[4];
    acc[0] = {}; acc[1] = {}; acc[2] = {}; acc[3] = {};

    // A fragment: row M = laneLo, K halves per ISA 16-bit 16x32 layout
    const _Float16* Arow = As + (size_t)(wo * 16 + laneLo) * K + laneHi * 8;
    // B fragment: column N = laneLo, K halves per ISA 16-bit 32x16 layout (same as A)
    const _Float16* Bbase = X + ((size_t)b * N + nwb + laneLo) * K + laneHi * 8;

    for (int k0 = 0; k0 < K; k0 += 32) {
        union { v16h v; uint4 q[2]; } a;
        a.q[0] = *(const uint4*)(Arow + k0);        // K = k0 + {0..7}  (+8 for hi lanes)
        a.q[1] = *(const uint4*)(Arow + k0 + 16);   // K = k0 + {16..23} (+8)
#pragma unroll
        for (int t = 0; t < 4; ++t) {
            union { v16h v; uint4 q[2]; } bf;
            const _Float16* bp = Bbase + (size_t)t * 16 * K + k0;
            bf.q[0] = *(const uint4*)bp;
            bf.q[1] = *(const uint4*)(bp + 16);
            acc[t] = __builtin_amdgcn_wmma_f32_16x16x32_f16(
                false, a.v, false, bf.v, (short)0, acc[t], false, false);
        }
    }

    float* hb = hpre + (size_t)b * H * N;
#pragma unroll
    for (int j = 0; j < 8; ++j) {
        const int o = obase + wo * 16 + laneHi * 8 + j;
        const float bo = bias[o];
        float psum = 0.f, psq = 0.f;
#pragma unroll
        for (int t = 0; t < 4; ++t) {
            const int n = nwb + t * 16 + laneLo;
            const float v = acc[t][j] + bo;
            hb[(size_t)o * N + n] = v;
            psum += v; psq += v * v;
        }
#pragma unroll
        for (int off = 8; off >= 1; off >>= 1) {
            psum += __shfl_xor(psum, off, 16);
            psq  += __shfl_xor(psq,  off, 16);
        }
        if (laneLo == 0) {
            atomicAdd(&sum[o], psum);
            atomicAdd(&sq[o],  psq);
        }
    }
}

// -------------------- BN1 + ReLU, transpose to [B][N][H] f16 for the second GEMM
__global__ __launch_bounds__(256) void bn1_kernel(const float* __restrict__ hpre,
                                                  const float* __restrict__ sum,
                                                  const float* __restrict__ sq,
                                                  const float* __restrict__ g,
                                                  const float* __restrict__ be,
                                                  _Float16* __restrict__ h1h) {
    __shared__ float t[32][33];
    const float invC = 1.f / (float)((size_t)B * N);
    const int b = blockIdx.z;
    const int n0 = blockIdx.x * 32, o0 = blockIdx.y * 32;
    const int tx = threadIdx.x, ty = threadIdx.y;
    for (int r = 0; r < 4; ++r) {
        const int o = o0 + ty + 8 * r;
        const float mean = sum[o] * invC;
        const float var  = sq[o] * invC - mean * mean;
        const float sc = g[o] * rsqrtf(var + EPS_BN);
        const float sh = be[o] - mean * sc;
        const float v = hpre[((size_t)b * H + o) * N + n0 + tx] * sc + sh;
        t[ty + 8 * r][tx] = fmaxf(v, 0.f);
    }
    __syncthreads();
    for (int r = 0; r < 4; ++r)
        h1h[((size_t)b * N + n0 + ty + 8 * r) * H + o0 + tx] = (_Float16)t[tx][ty + 8 * r];
}

// ---------------------------------- BN2 + ReLU -> output [B,H,N] f32 (elementwise)
__global__ __launch_bounds__(256) void bn2_kernel(const float* __restrict__ hpre,
                                                  const float* __restrict__ sum,
                                                  const float* __restrict__ sq,
                                                  const float* __restrict__ g,
                                                  const float* __restrict__ be,
                                                  float* __restrict__ out) {
    const float invC = 1.f / (float)((size_t)B * N);
    const size_t idx = (size_t)blockIdx.x * 256 + threadIdx.x;
    const int o = (int)((idx >> 12) & (H - 1));     // N = 4096 = 2^12
    const float mean = sum[o] * invC;
    const float var  = sq[o] * invC - mean * mean;
    const float sc = g[o] * rsqrtf(var + EPS_BN);
    const float sh = be[o] - mean * sc;
    out[idx] = fmaxf(hpre[idx] * sc + sh, 0.f);
}

// ---------------------------------------------------------------------------------
extern "C" void kernel_launch(void* const* d_in, const int* in_sizes, int n_in,
                              void* d_out, int out_size, void* d_ws, size_t ws_size,
                              hipStream_t stream) {
    (void)in_sizes; (void)n_in; (void)out_size; (void)ws_size;
    const float* pc1   = (const float*)d_in[0];
    const float* pc2   = (const float*)d_in[1];
    const float* feat1 = (const float*)d_in[2];
    const float* feat2 = (const float*)d_in[3];
    const float* w1    = (const float*)d_in[4];
    const float* b1    = (const float*)d_in[5];
    const float* g1    = (const float*)d_in[6];
    const float* be1   = (const float*)d_in[7];
    const float* w2    = (const float*)d_in[8];
    const float* b2    = (const float*)d_in[9];
    const float* g2    = (const float*)d_in[10];
    const float* be2   = (const float*)d_in[11];
    float* out = (float*)d_out;

    char* ws = (char*)d_ws;
    size_t off = 0;
    float*    f2t  = (float*)(ws + off);    off += (size_t)B * M * C2 * 4;   // 16.8 MB
    int*      idx3 = (int*)(ws + off);      off += (size_t)B * N * 3 * 4;    // 0.8 MB
    float*    w3   = (float*)(ws + off);    off += (size_t)B * N * 3 * 4;    // 0.8 MB
    _Float16* Xh   = (_Float16*)(ws + off); off += (size_t)B * N * K1 * 2;   // 67 MB
    float*    hpre = (float*)(ws + off);    off += (size_t)B * H * N * 4;    // 67 MB (reused by both layers)
    _Float16* h1h  = (_Float16*)(ws + off); off += (size_t)B * N * H * 2;    // 33.5 MB
    float*    stats = (float*)(ws + off);   off += 4 * H * 4;
    float* sum1 = stats;          float* sq1 = stats + H;
    float* sum2 = stats + 2 * H;  float* sq2 = stats + 3 * H;

    hipMemsetAsync(stats, 0, 4 * H * sizeof(float), stream);

    knn_kernel<<<dim3(N / 256, B), 256, 0, stream>>>(pc1, pc2, idx3, w3);
    f2_transpose<<<dim3(M / 32, C2 / 32, B), dim3(32, 8), 0, stream>>>(feat2, f2t);
    f1_transpose<<<dim3(N / 32, C1 / 32, B), dim3(32, 8), 0, stream>>>(feat1, Xh);
    interp_kernel<<<dim3(N / 8, B), 256, 0, stream>>>(f2t, idx3, w3, Xh);

    gemm_kernel<K1><<<dim3(N / 256, H / 32, B), 256, 0, stream>>>(Xh, w1, b1, hpre, sum1, sq1);
    bn1_kernel<<<dim3(N / 32, H / 32, B), dim3(32, 8), 0, stream>>>(hpre, sum1, sq1, g1, be1, h1h);
    gemm_kernel<H><<<dim3(N / 256, H / 32, B), 256, 0, stream>>>(h1h, w2, b2, hpre, sum2, sq2);
    bn2_kernel<<<dim3((unsigned)((size_t)B * H * N / 256)), 256, 0, stream>>>(hpre, sum2, sq2, g2, be2, out);
}
